// BlendShapesModel_23708219474158
// MI455X (gfx1250) — matrix operand order repaired and checked by reading, not compiled, not run on hardware
//
#include <hip/hip_runtime.h>

// Problem constants
#define N_VERT  6890
#define N_JOINT 23
#define BPJ     8
#define BATCH   1024
#define KDIM    184          // N_JOINT * BPJ
#define KP      192          // padded K: 6 * 32
#define NKT     6            // number of 32-wide K steps
#define NDIM    20670        // N_VERT * 3
#define NTILE   1292         // ceil(NDIM / 16)
#define MT      64           // BATCH / 16
#define NBLK    4            // n-tiles per wave
#define NB      323          // NTILE / NBLK (exact)

typedef __attribute__((ext_vector_type(16))) __bf16 v16bf;
typedef __attribute__((ext_vector_type(8)))  float  v8f;

// round-to-nearest-even f32 -> bf16 (raw bits)
__device__ __forceinline__ unsigned short f2bf(float f) {
  union { float f; unsigned int u; } x; x.f = f;
  unsigned int u = x.u;
  unsigned int r = u + 0x7FFFu + ((u >> 16) & 1u);
  return (unsigned short)(r >> 16);
}

// Swizzled A-fragment element offset (ushort units) for logical A[m][k].
// Matches ISA "16-bit A-Matrix 16x32": lane = 16*h + (m%16), v16 element e with
//   K = 8h+e (e<8) or 8+8h+e (e>=8); fragments stored contiguously per (mt,kt,lane).
__device__ __forceinline__ int a_index(int m, int k) {
  int mt = m >> 4, M = m & 15;
  int kt = k >> 5, kk = k & 31;
  int h, e;
  if      (kk <  8) { h = 0; e = kk;      }
  else if (kk < 16) { h = 1; e = kk - 8;  }
  else if (kk < 24) { h = 0; e = kk - 8;  }
  else              { h = 1; e = kk - 16; }
  int lane = (h << 4) | M;
  return ((mt * NKT + kt) * 32 + lane) * 16 + e;
}

// ---------------------------------------------------------------------------
// Kernel 1: per-joint MLP [9 -> 18 -> 32 -> 8], weights in LDS.
// grid (23, BATCH/256), block 256. Writes coff into swizzled bf16 A buffer.
// ---------------------------------------------------------------------------
__global__ __launch_bounds__(256)
void coff_mlp_kernel(const float* __restrict__ pose,
                     const float* __restrict__ w1, const float* __restrict__ b1,
                     const float* __restrict__ w2, const float* __restrict__ b2,
                     const float* __restrict__ w3, const float* __restrict__ b3,
                     unsigned short* __restrict__ A_swz) {
  __shared__ float lds[162 + 18 + 576 + 32 + 256 + 8];   // 1052 floats
  const int j   = blockIdx.x;
  const int tid = threadIdx.x;
  float* lw1 = lds;        float* lb1 = lw1 + 162;
  float* lw2 = lb1 + 18;   float* lb2 = lw2 + 576;
  float* lw3 = lb2 + 32;   float* lb3 = lw3 + 256;
  for (int i = tid; i < 162; i += 256) lw1[i] = w1[j * 162 + i];
  for (int i = tid; i < 18;  i += 256) lb1[i] = b1[j * 18  + i];
  for (int i = tid; i < 576; i += 256) lw2[i] = w2[j * 576 + i];
  for (int i = tid; i < 32;  i += 256) lb2[i] = b2[j * 32  + i];
  for (int i = tid; i < 256; i += 256) lw3[i] = w3[j * 256 + i];
  for (int i = tid; i < 8;   i += 256) lb3[i] = b3[j * 8   + i];
  __syncthreads();

  const int b = blockIdx.y * 256 + tid;

  // pose_repr = pose[b, j+1, :, :].flatten() - eye(3).flatten()
  float pr[9];
  #pragma unroll
  for (int i = 0; i < 9; ++i) {
    float e = (i == 0 || i == 4 || i == 8) ? 1.0f : 0.0f;
    pr[i] = pose[(size_t)b * ((N_JOINT + 1) * 9) + (j + 1) * 9 + i] - e;
  }

  float h1[18];
  #pragma unroll
  for (int o = 0; o < 18; ++o) {
    float s = lb1[o];
    #pragma unroll
    for (int i = 0; i < 9; ++i) s += pr[i] * lw1[i * 18 + o];
    h1[o] = fmaxf(s, 0.0f);
  }
  float h2[32];
  #pragma unroll
  for (int o = 0; o < 32; ++o) {
    float s = lb2[o];
    #pragma unroll
    for (int i = 0; i < 18; ++i) s += h1[i] * lw2[i * 32 + o];
    h2[o] = fmaxf(s, 0.0f);
  }
  #pragma unroll
  for (int o = 0; o < 8; ++o) {
    float s = lb3[o];
    #pragma unroll
    for (int i = 0; i < 32; ++i) s += h2[i] * lw3[i * 8 + o];
    A_swz[a_index(b, j * BPJ + o)] = f2bf(s);
  }
  // zero-pad K = 184..191 (done once, by the j==0 blocks)
  if (j == 0) {
    #pragma unroll
    for (int k = KDIM; k < KP; ++k) A_swz[a_index(b, k)] = 0;
  }
}

// ---------------------------------------------------------------------------
// Kernel 2: build masked-basis B matrix (K x N), bf16, in WMMA B-fragment
// layout. Each thread emits one 16-element (32B) contiguous fragment slice:
// lane = 16*h + (n%16); element e corresponds to k = 32*kt + 8*h + e (e<8), else 32*kt + 8 + 8*h + e.
// ---------------------------------------------------------------------------
__global__ __launch_bounds__(256)
void basis_swz_kernel(const float* __restrict__ basis,
                      const float* __restrict__ mask,
                      unsigned short* __restrict__ B_swz) {
  int id = blockIdx.x * 256 + threadIdx.x;      // [0, NTILE*NKT*32)
  if (id >= NTILE * NKT * 32) return;
  int lane = id & 31;
  int rest = id >> 5;
  int kt = rest % NKT;
  int nt = rest / NKT;
  int h = lane >> 4, Nl = lane & 15;
  int n = nt * 16 + Nl;

  unsigned short tmp[16];
  if (n < NDIM) {
    int v = n / 3, c = n % 3;
    #pragma unroll
    for (int e = 0; e < 16; ++e) {
      int k = kt * 32 + (e < 8 ? 8 * h + e : 16 + 8 * h + (e - 8));
      float val = 0.0f;
      if (k < KDIM) {
        int jj = k >> 3, p = k & 7;
        val = basis[(v * BPJ + p) * 3 + c] * mask[v * N_JOINT + jj];
      }
      tmp[e] = f2bf(val);
    }
  } else {
    #pragma unroll
    for (int e = 0; e < 16; ++e) tmp[e] = 0;
  }
  uint4* dst = (uint4*)(B_swz + (size_t)id * 16);
  dst[0] = *(uint4*)(tmp);
  dst[1] = *(uint4*)(tmp + 8);
}

// ---------------------------------------------------------------------------
// Kernel 3: WMMA GEMM. One wave per 16(M) x 64(N) tile.
// A/B fragments are pre-swizzled -> each lane loads 32 contiguous bytes.
// 6 k-steps x 4 n-tiles = 24 v_wmma_f32_16x16x32_bf16 per wave.
// Grid covers exactly MT*NB waves, so EXEC is all-ones at every WMMA.
// ---------------------------------------------------------------------------
__global__ __launch_bounds__(256)
void wmma_gemm_kernel(const unsigned short* __restrict__ A_swz,
                      const unsigned short* __restrict__ B_swz,
                      float* __restrict__ out) {
  int gtid = blockIdx.x * 256 + threadIdx.x;
  int wave = gtid >> 5;
  int lane = gtid & 31;
  int mt = wave / NB;
  int nb = wave % NB;

  v8f acc[NBLK];
  #pragma unroll
  for (int i = 0; i < NBLK; ++i) {
    #pragma unroll
    for (int e = 0; e < 8; ++e) acc[i][e] = 0.0f;
  }

  #pragma unroll
  for (int kt = 0; kt < NKT; ++kt) {
    v16bf a = *(const v16bf*)(A_swz + ((size_t)(mt * NKT + kt) * 32 + lane) * 16);
    #pragma unroll
    for (int i = 0; i < NBLK; ++i) {
      int nt = nb * NBLK + i;
      v16bf bm = *(const v16bf*)(B_swz + ((size_t)(nt * NKT + kt) * 32 + lane) * 16);
      acc[i] = __builtin_amdgcn_wmma_f32_16x16x32_bf16(
          /*neg_a=*/false, a, /*neg_b=*/false, bm,
          /*c_mod=*/(short)0, acc[i], /*reuse_a=*/false, /*reuse_b=*/false);
    }
  }

  int h = lane >> 4, Nl = lane & 15;
  #pragma unroll
  for (int i = 0; i < NBLK; ++i) {
    int n = (nb * NBLK + i) * 16 + Nl;
    if (n < NDIM) {
      #pragma unroll
      for (int r = 0; r < 8; ++r) {
        int m = mt * 16 + r + 8 * h;           // C/D layout: VGPR r -> M = r + 8h
        out[(size_t)m * NDIM + n] = acc[i][r];
      }
    }
  }
}

extern "C" void kernel_launch(void* const* d_in, const int* in_sizes, int n_in,
                              void* d_out, int out_size, void* d_ws, size_t ws_size,
                              hipStream_t stream) {
  const float* pose  = (const float*)d_in[0];
  const float* basis = (const float*)d_in[1];
  const float* mask  = (const float*)d_in[2];
  const float* w1 = (const float*)d_in[3];
  const float* b1 = (const float*)d_in[4];
  const float* w2 = (const float*)d_in[5];
  const float* b2 = (const float*)d_in[6];
  const float* w3 = (const float*)d_in[7];
  const float* b3 = (const float*)d_in[8];
  float* out = (float*)d_out;

  // ws: [A_swz: 64*6*512*2 = 384KB][pad to 512KB][B_swz: 1292*6*512*2 ~= 7.9MB]
  unsigned short* A_swz = (unsigned short*)d_ws;
  unsigned short* B_swz = (unsigned short*)((char*)d_ws + (size_t)512 * 1024);

  coff_mlp_kernel<<<dim3(N_JOINT, BATCH / 256), 256, 0, stream>>>(
      pose, w1, b1, w2, b2, w3, b3, A_swz);

  int btot = NTILE * NKT * 32;                      // 248064 threads
  basis_swz_kernel<<<(btot + 255) / 256, 256, 0, stream>>>(basis, mask, B_swz);

  int nthreads = MT * NB * 32;                      // 20672 waves exactly
  wmma_gemm_kernel<<<nthreads / 256, 256, 0, stream>>>(A_swz, B_swz, out);
}
